// ADF_Other_Act_Layer_19387482375006
// MI455X (gfx1250) — compile-verified, matrix operand order checked by reading, not run on hardware
//
#include <hip/hip_runtime.h>
#include <math.h>

// ---------------------------------------------------------------------------
// ADF moment propagation of N(mu, sigma^2) through tanh via 128-pt
// Gauss-Hermite quadrature.
//
//   m1[n] = sum_q W_q * tanh(mu_n + sqrt(2)*sigma_n*x_q)
//   out   = [m1, m2 - m1^2],  m2 = sum_q W_q * tanh(...)^2
//
// TRANS-bound kernel (128 v_tanh_f32 per element). The weighted reductions
// over q are offloaded to the matrix pipe with v_wmma_f32_16x16x32_f16 using
// A == ones so D = column-sum of B (K-layout-permutation invariant).
// GH nodes/weights are generated on-device (Sturm bisection + Newton).
// ---------------------------------------------------------------------------

typedef __attribute__((ext_vector_type(16))) _Float16 v16h;
typedef __attribute__((ext_vector_type(8)))  float    v8f;

#define GH_Q 128

__device__ __align__(16) float g_nodes[GH_Q];   // x_q (ascending)
__device__ __align__(16) float g_wts[GH_Q];     // w_q / sqrt(pi), sums to 1

// ---------------------------------------------------------------------------
// Setup: 128 threads, one Gauss-Hermite root each.
// Orthonormal Hermite recurrence (weight e^{-x^2}):
//   h_0 = pi^{-1/4};  h_j = x*sqrt(2/j)*h_{j-1} - sqrt((j-1)/j)*h_{j-2}
// #roots of h_128 below lam == 128 - (#sign changes of h_0..h_128 at lam).
// Bisection (f32, division-free, rescaled) then Newton polish in f64 using
// h_128'(z) = sqrt(256)*h_127(z).  Weight: w = 2/pp^2 (NR gauher), /sqrt(pi).
// ---------------------------------------------------------------------------
__global__ __launch_bounds__(GH_Q) void gh_setup_kernel() {
  __shared__ double c1d[GH_Q + 1], c2d[GH_Q + 1];
  __shared__ float  c1f[GH_Q + 1], c2f[GH_Q + 1];
  const int k = threadIdx.x;                       // root index 0..127
  {
    const int j = k + 1;                           // 1..128
    const double a = sqrt(2.0 / (double)j);
    const double b = sqrt(((double)j - 1.0) / (double)j);
    c1d[j] = a;  c2d[j] = b;
    c1f[j] = (float)a;  c2f[j] = (float)b;
  }
  __syncthreads();

  // --- bisection on Sturm sign-change count (f32) ---
  float lo = -16.5f, hi = 16.5f;                   // spectrum in (-sqrt(257), sqrt(257))
  for (int it = 0; it < 34; ++it) {
    const float lam = 0.5f * (lo + hi);
    float pm1 = 0.75112554446494248f;              // pi^(-1/4) = h_0 > 0
    float pm2 = 0.0f;
    int changes = 0;
    bool prevneg = false;
    for (int j = 1; j <= GH_Q; ++j) {
      float p = lam * c1f[j] * pm1 - c2f[j] * pm2;
      const bool neg = (p < 0.0f);
      changes += (neg != prevneg) ? 1 : 0;
      prevneg = neg;
      // rescale to avoid f32 overflow outside the oscillatory region
      const float s = (fabsf(p) > 1e18f) ? 1e-18f : 1.0f;
      pm2 = pm1 * s;
      pm1 = p * s;
    }
    const int nless = GH_Q - changes;              // #roots < lam
    if (nless > k) hi = lam; else lo = lam;
  }

  // --- Newton polish (f64); final pass evaluates pp at converged z ---
  double z  = 0.5 * ((double)lo + (double)hi);
  double pp = 1.0;
  for (int it = 0; it < 4; ++it) {
    double pm1 = 0.7511255444649424828587;         // pi^(-1/4)
    double pm2 = 0.0;
    for (int j = 1; j <= GH_Q; ++j) {
      const double p = z * c1d[j] * pm1 - c2d[j] * pm2;
      pm2 = pm1;  pm1 = p;
    }
    pp = 16.0 * pm2;                               // sqrt(2*128) * h_127(z)
    if (it < 3) z -= pm1 / pp;
  }
  const double w = 2.0 / (pp * pp);                // GH weight (sum = sqrt(pi))
  g_nodes[k] = (float)z;
  g_wts[k]   = (float)(w / 1.7724538509055160273); // normalize: sum -> 1
}

// ---------------------------------------------------------------------------
__device__ __forceinline__ float fast_tanh(float x) {
#if __has_builtin(__builtin_amdgcn_tanhf)
  return __builtin_amdgcn_tanhf(x);                // gfx1250 v_tanh_f32
#elif __has_builtin(__builtin_amdgcn_tanh_f32)
  return __builtin_amdgcn_tanh_f32(x);
#elif __has_builtin(__builtin_amdgcn_exp2f) && __has_builtin(__builtin_amdgcn_rcpf)
  // tanh(x) = 1 - 2/(exp(2x)+1);  exp(2x) = 2^(2*log2(e)*x)
  const float e = __builtin_amdgcn_exp2f(2.8853900817779268f * x);
  return 1.0f - 2.0f * __builtin_amdgcn_rcpf(e + 1.0f);
#else
  const float e = __expf(2.0f * x);
  return 1.0f - 2.0f / (e + 1.0f);
#endif
}

// ---------------------------------------------------------------------------
// Main kernel: one wave -> 16 output columns.
// B (32x16 f16): lane L supplies column (L&15); half-waves supply the two
// K=16 halves of each K=32 chunk (q = c*32 + (L&16) + e).  A = ones, so
// D[m][n] = sum_k B[k][n] for every m -> result identical in both half-waves.
// ---------------------------------------------------------------------------
__global__ __launch_bounds__(256) void adf_tanh_gh_kernel(
    const float* __restrict__ X, float* __restrict__ out, const int N) {
  const int lane = threadIdx.x & 31;
  const int wave = threadIdx.x >> 5;
  const long long tile = (long long)blockIdx.x * 8 + wave;
  const int col  = (int)(tile * 16) + (lane & 15);
  const int colc = (col < N) ? col : (N - 1);      // clamp: keep EXEC all-ones

  const float mean = X[colc];
  const float var  = X[(size_t)N + colc];
  const float s2s  = sqrtf(var) * 1.4142135623730951f;   // sqrt(2)*sigma

  const int off = (lane & 16);                     // K-offset of this half-wave
  const _Float16 one = (_Float16)1.0f;
  const v16h ones = {one, one, one, one, one, one, one, one,
                     one, one, one, one, one, one, one, one};
  v8f acc1 = {};                                   // accumulates m1 per column
  v8f acc2 = {};                                   // accumulates m2 per column

#pragma unroll
  for (int c = 0; c < 4; ++c) {                    // 4 chunks of K=32 -> Q=128
    const int q0 = c * 32 + off;
    const float4* np = (const float4*)(g_nodes + q0);
    const float4* wp = (const float4*)(g_wts + q0);
    v16h b1, b2;
#pragma unroll
    for (int v = 0; v < 4; ++v) {
      const float4 nd = np[v];
      const float4 wt = wp[v];
      const float a0 = fast_tanh(fmaf(s2s, nd.x, mean));
      const float a1 = fast_tanh(fmaf(s2s, nd.y, mean));
      const float a2 = fast_tanh(fmaf(s2s, nd.z, mean));
      const float a3 = fast_tanh(fmaf(s2s, nd.w, mean));
      const float p0 = wt.x * a0, p1 = wt.y * a1;
      const float p2 = wt.z * a2, p3 = wt.w * a3;
      b1[4 * v + 0] = (_Float16)p0;  b2[4 * v + 0] = (_Float16)(p0 * a0);
      b1[4 * v + 1] = (_Float16)p1;  b2[4 * v + 1] = (_Float16)(p1 * a1);
      b1[4 * v + 2] = (_Float16)p2;  b2[4 * v + 2] = (_Float16)(p2 * a2);
      b1[4 * v + 3] = (_Float16)p3;  b2[4 * v + 3] = (_Float16)(p3 * a3);
    }
    // D = ones * B + C  ==  per-column K-sum (permutation-invariant in K)
    acc1 = __builtin_amdgcn_wmma_f32_16x16x32_f16(false, ones, false, b1,
                                                  (short)0, acc1, false, false);
    acc2 = __builtin_amdgcn_wmma_f32_16x16x32_f16(false, ones, false, b2,
                                                  (short)0, acc2, false, false);
  }

  if (col < N) {
    const float m1 = acc1[0];                      // all D rows identical
    const float m2 = acc2[0];
    if (lane < 16) out[col] = m1;                  // halves hold same data:
    else           out[(size_t)N + col] = m2 - m1 * m1;  // split the stores
  }
}

// ---------------------------------------------------------------------------
extern "C" void kernel_launch(void* const* d_in, const int* in_sizes, int n_in,
                              void* d_out, int out_size, void* d_ws,
                              size_t ws_size, hipStream_t stream) {
  (void)n_in; (void)d_ws; (void)ws_size; (void)out_size;
  const float* X = (const float*)d_in[0];          // [2, N] flat
  float* out = (float*)d_out;                      // [2, N] flat
  const int N = in_sizes[0] / 2;

  gh_setup_kernel<<<1, GH_Q, 0, stream>>>();

  const int tiles  = (N + 15) / 16;                // 16 columns per wave
  const int blocks = (tiles + 7) / 8;              // 8 waves per block
  adf_tanh_gh_kernel<<<blocks, 256, 0, stream>>>(X, out, N);
}
